// STGCNModel_19275813224639
// MI455X (gfx1250) — matrix operand order repaired and checked by reading, not compiled, not run on hardware
//
#include <hip/hip_runtime.h>
#include <cstdint>

#define N_NODES 10000
#define T_STEPS 12
#define BATCH   8
#define N_EDGES 160000
#define HID     64

typedef __attribute__((ext_vector_type(16))) __bf16 v16bf;
typedef __attribute__((ext_vector_type(8)))  __bf16 v8bf;
typedef __attribute__((ext_vector_type(8)))  float  v8f;

__device__ __forceinline__ __bf16 f2bf(float x) {
  union { float f; uint32_t u; } v; v.f = x;
  uint32_t r = v.u + 0x7FFFu + ((v.u >> 16) & 1u);   // round-to-nearest-even
  uint16_t h = (uint16_t)(r >> 16);
  return __builtin_bit_cast(__bf16, h);
}

// ---------------------------------------------------------------------------
// Small elementwise / graph kernels
// ---------------------------------------------------------------------------

__global__ void fill_kernel(float* __restrict__ p, float val, int n) {
  int t = blockIdx.x * blockDim.x + threadIdx.x;
  if (t < n) p[t] = val;
}

__global__ void deg_scatter(const int* __restrict__ dst, const float* __restrict__ w,
                            float* __restrict__ deg, int E) {
  int e = blockIdx.x * blockDim.x + threadIdx.x;
  if (e < E) atomicAdd(deg + dst[e], w[e]);
}

__global__ void selfnorm_kernel(const float* __restrict__ deg, float* __restrict__ selfn, int n) {
  int t = blockIdx.x * blockDim.x + threadIdx.x;
  if (t < n) selfn[t] = 1.0f / deg[t];           // dinv^2 (deg >= 1 from self loop)
}

__global__ void edge_norm_kernel(const int* __restrict__ src, const int* __restrict__ dst,
                                 const float* __restrict__ w, const float* __restrict__ deg,
                                 float* __restrict__ norm, int E) {
  int e = blockIdx.x * blockDim.x + threadIdx.x;
  if (e < E) norm[e] = rsqrtf(deg[src[e]]) * w[e] * rsqrtf(deg[dst[e]]);
}

// h0 = X[b,t] @ W1  (N x 2) @ (2 x 64)
__global__ void xw1_kernel(const float* __restrict__ X, const float* __restrict__ W1,
                           float* __restrict__ out, int xoff) {
  int t = blockIdx.x * blockDim.x + threadIdx.x;
  if (t >= N_NODES * HID) return;
  int n = t >> 6, c = t & 63;
  float x0 = X[xoff + n * 2 + 0], x1 = X[xoff + n * 2 + 1];
  out[t] = x0 * W1[c] + x1 * W1[HID + c];
}

// init aggregation buffer with the self-loop term: hout = selfn[n] * pre
__global__ void selfloop_init(const float* __restrict__ pre, const float* __restrict__ selfn,
                              float* __restrict__ hout) {
  int t = blockIdx.x * blockDim.x + threadIdx.x;
  if (t >= N_NODES * HID) return;
  int n = t >> 6;
  hout[t] = selfn[n] * pre[t];
}

// out[dst] += norm * hin[src]  (16 threads per edge, 4 channels each)
__global__ void scatter_agg(const int* __restrict__ src, const int* __restrict__ dst,
                            const float* __restrict__ norm,
                            const float* __restrict__ hin, float* __restrict__ hout, int E) {
  int t = blockIdx.x * blockDim.x + threadIdx.x;
  int e = t >> 4;
  if (e >= E) return;
  int c4 = (t & 15) << 2;
  int s = src[e], d = dst[e];
  float w = norm[e];
  const float4 v = *(const float4*)(hin + (size_t)s * HID + c4);
  atomicAdd(hout + (size_t)d * HID + c4 + 0, w * v.x);
  atomicAdd(hout + (size_t)d * HID + c4 + 1, w * v.y);
  atomicAdd(hout + (size_t)d * HID + c4 + 2, w * v.z);
  atomicAdd(hout + (size_t)d * HID + c4 + 3, w * v.w);
}

// out = bf16(relu(agg + bias[c]))
__global__ void relu_bias_bf16(const float* __restrict__ agg, const float* __restrict__ bias,
                               __bf16* __restrict__ out) {
  int t = blockIdx.x * blockDim.x + threadIdx.x;
  if (t >= N_NODES * HID) return;
  int c = t & 63;
  float v = agg[t] + bias[c];
  out[t] = f2bf(v > 0.0f ? v : 0.0f);
}

// ---------------------------------------------------------------------------
// Weight repack into WMMA B-fragment layout:
//   dst[((s*4+ct)*32 + lane)*16 + j] = W[s*32 + (lane>>4)*16 + j][ct*16 + (lane&15)]
// ---------------------------------------------------------------------------

__global__ void pack_w2(const float* __restrict__ W, __bf16* __restrict__ dst) {
  int t = blockIdx.x * blockDim.x + threadIdx.x;       // 64*64 = 4096
  if (t >= HID * HID) return;
  int j = t & 15, lane = (t >> 4) & 31, frag = t >> 9;
  int ct = frag & 3, s = frag >> 2;
  int kk = s * 32 + (lane >> 4) * 16 + j;              // cin
  int n  = ct * 16 + (lane & 15);                      // cout
  dst[t] = f2bf(W[kk * HID + n]);                      // W2 layout (cin, cout)
}

// conv weights, src layout OIH: W[cout][cin][ktap] stride 3; kk = tap*64 + cin
__global__ void pack_convw(const float* __restrict__ W, __bf16* __restrict__ dst, int total) {
  int t = blockIdx.x * blockDim.x + threadIdx.x;
  if (t >= total) return;
  int j = t & 15, lane = (t >> 4) & 31, frag = t >> 9;
  int ct = frag & 3, s = frag >> 2;
  int kk = s * 32 + (lane >> 4) * 16 + j;
  int cout = ct * 16 + (lane & 15);
  int tap = kk >> 6, cin = kk & 63;
  dst[t] = f2bf(W[(cout * HID + cin) * 3 + tap]);
}

// ---------------------------------------------------------------------------
// Generic bf16 WMMA GEMM: out[M x 64] = A[M x 32*NK] @ Wpack
//   A sourced per 32-wide K-step from (pointer, column-offset); nullptr => zero pad.
//   Packed weights staged in LDS once per block; fragments read via ds_load.
//   blockIdx.y selects the batch (pointers advance by batch strides).
//   MODE 0: raw f32   MODE 1: relu(x+bias) -> bf16   MODE 2: relu(x+bias) -> f32
// ---------------------------------------------------------------------------

struct KSrcs {
  const __bf16* p[6];
  int coloff[6];
};

template<int NK, int MODE>
__global__ __launch_bounds__(256)
void gemm_bf16_wmma(KSrcs ks, const __bf16* __restrict__ Wpack,
                    const float* __restrict__ bias,
                    float* __restrict__ outf, __bf16* __restrict__ outbf,
                    int M, long strideIn, long strideOut) {
  __shared__ __bf16 wlds[NK * 4 * 32 * 16];            // NK*4 fragments, <= 24 KB

  // cooperative stage of packed weights into LDS (16B chunks)
  {
    const uint4* s4 = (const uint4*)Wpack;
    uint4*       d4 = (uint4*)wlds;
#pragma unroll
    for (int i = 0; i < NK; ++i) d4[i * 256 + threadIdx.x] = s4[i * 256 + threadIdx.x];
  }
  __syncthreads();

  const int lane  = threadIdx.x & 31;
  const int wave  = threadIdx.x >> 5;
  const int rowBase = blockIdx.x * 128 + wave * 16;    // 8 waves x 16 rows
  const int half  = lane >> 4;
  const int mlane = lane & 15;
  const long boffIn  = (long)blockIdx.y * strideIn;
  const long boffOut = (long)blockIdx.y * strideOut;

  v8f acc[4];
#pragma unroll
  for (int ct = 0; ct < 4; ++ct)
#pragma unroll
    for (int i = 0; i < 8; ++i) acc[ct][i] = 0.0f;

#pragma unroll
  for (int s = 0; s < NK; ++s) {
    v16bf a;
    const __bf16* p = ks.p[s];
    if (p) {
      int r = rowBase + mlane;
      if (r >= M) r = M - 1;
      const __bf16* rp = p + boffIn + (size_t)r * HID + ks.coloff[s];
      // A in this code's contiguous k order (not the ISA table order): lane half h holds K = h*16+0..15 (matches packed B)
      a = *(const v16bf*)(rp + half * 16);
      __builtin_prefetch(rp + 128 * HID, 0, 0);        // next row-tile -> global_prefetch_b8
    } else {
      const __bf16 bz = __builtin_bit_cast(__bf16, (uint16_t)0);
#pragma unroll
      for (int i = 0; i < 16; ++i) a[i] = bz;
    }
#pragma unroll
    for (int ct = 0; ct < 4; ++ct) {
      v16bf bb = *(const v16bf*)(wlds + ((s * 4 + ct) * 32 + lane) * 16);
      acc[ct] = __builtin_amdgcn_wmma_f32_16x16x32_bf16(
          false, a, false, bb, (short)0, acc[ct], false, false);
    }
  }

#pragma unroll
  for (int ct = 0; ct < 4; ++ct) {
#pragma unroll
    for (int r = 0; r < 8; ++r) {
      int m = rowBase + half * 8 + r;                  // C/D layout: vgpr r, lane half
      if (m < M) {
        int n = ct * 16 + mlane;
        float v = acc[ct][r];
        if (MODE == 0) {
          outf[boffOut + (size_t)m * HID + n] = v;
        } else {
          v += bias[n];
          v = v > 0.0f ? v : 0.0f;
          if (MODE == 1) outbf[boffOut + (size_t)m * HID + n] = f2bf(v);
          else           outf [boffOut + (size_t)m * HID + n] = v;
        }
      }
    }
  }
}

// final projection: out[b*N+n] = h2f[b,n,:] . out_w + out_b
__global__ void proj_kernel(const float* __restrict__ h2f, const float* __restrict__ outw,
                            const float* __restrict__ outb, float* __restrict__ out, int total) {
  int t = blockIdx.x * blockDim.x + threadIdx.x;
  if (t >= total) return;
  const float* r = h2f + (size_t)t * HID;
  float s = 0.0f;
#pragma unroll
  for (int c = 0; c < HID; ++c) s += r[c] * outw[c];
  out[t] = s + outb[0];
}

// ---------------------------------------------------------------------------

static inline int cdiv(int a, int b) { return (a + b - 1) / b; }

extern "C" void kernel_launch(void* const* d_in, const int* in_sizes, int n_in,
                              void* d_out, int out_size, void* d_ws, size_t ws_size,
                              hipStream_t stream) {
  (void)in_sizes; (void)n_in; (void)out_size; (void)ws_size;

  const float* X    = (const float*)d_in[0];
  const int*   ei   = (const int*)  d_in[1];
  const float* ew   = (const float*)d_in[2];
  const float* W1   = (const float*)d_in[3];
  const float* b1   = (const float*)d_in[4];
  const float* W2   = (const float*)d_in[5];
  const float* b2   = (const float*)d_in[6];
  const float* tc1w = (const float*)d_in[7];
  const float* tc1b = (const float*)d_in[8];
  const float* tc2w = (const float*)d_in[9];
  const float* tc2b = (const float*)d_in[10];
  const float* outw = (const float*)d_in[11];
  const float* outb = (const float*)d_in[12];
  float* out = (float*)d_out;

  const int* srcIdx = ei;
  const int* dstIdx = ei + N_EDGES;

  char* ws = (char*)d_ws;
  size_t off = 0;
  auto alloc = [&](size_t bytes) -> char* {
    char* p = ws + off;
    off += (bytes + 255) & ~(size_t)255;
    return p;
  };

  float*  deg   = (float*) alloc((size_t)N_NODES * 4);
  float*  selfn = (float*) alloc((size_t)N_NODES * 4);
  float*  norm  = (float*) alloc((size_t)N_EDGES * 4);
  float*  tmpA  = (float*) alloc((size_t)N_NODES * HID * 4);
  float*  tmpB  = (float*) alloc((size_t)N_NODES * HID * 4);
  __bf16* h1bf  = (__bf16*)alloc((size_t)N_NODES * HID * 2);
  __bf16* spat  = (__bf16*)alloc((size_t)BATCH * 3 * N_NODES * HID * 2);
  __bf16* c1    = (__bf16*)alloc((size_t)BATCH * 2 * N_NODES * HID * 2);
  float*  h2f   = (float*) alloc((size_t)BATCH * N_NODES * HID * 4);
  __bf16* W2p   = (__bf16*)alloc((size_t)HID * HID * 2);
  __bf16* Wc1p  = (__bf16*)alloc((size_t)3 * HID * HID * 2);
  __bf16* Wc2p  = (__bf16*)alloc((size_t)2 * HID * HID * 2);

  const int thr = 256;
  const int gemmGrid = cdiv(N_NODES, 128);
  const long NH = (long)N_NODES * HID;

  // --- GCN normalization (self-loop weight 1 folded into deg init) ---
  fill_kernel    <<<cdiv(N_NODES, thr), thr, 0, stream>>>(deg, 1.0f, N_NODES);
  deg_scatter    <<<cdiv(N_EDGES, thr), thr, 0, stream>>>(dstIdx, ew, deg, N_EDGES);
  selfnorm_kernel<<<cdiv(N_NODES, thr), thr, 0, stream>>>(deg, selfn, N_NODES);
  edge_norm_kernel<<<cdiv(N_EDGES, thr), thr, 0, stream>>>(srcIdx, dstIdx, ew, deg, norm, N_EDGES);

  // --- weight repacks to WMMA fragment layout ---
  pack_w2   <<<cdiv(HID * HID, thr), thr, 0, stream>>>(W2, W2p);
  pack_convw<<<cdiv(3 * HID * HID, thr), thr, 0, stream>>>(tc1w, Wc1p, 3 * HID * HID);
  pack_convw<<<cdiv(2 * HID * HID, thr), thr, 0, stream>>>(tc2w, Wc2p, 2 * HID * HID);

  // --- spatial GCN, only the 3 timesteps that reach the final output ---
  for (int b = 0; b < BATCH; ++b) {
    for (int ti = 0; ti < 3; ++ti) {
      int t = T_STEPS - 3 + ti;                        // 9,10,11
      int xoff = ((b * T_STEPS) + t) * N_NODES * 2;
      xw1_kernel   <<<cdiv(N_NODES * HID, thr), thr, 0, stream>>>(X, W1, tmpA, xoff);
      selfloop_init<<<cdiv(N_NODES * HID, thr), thr, 0, stream>>>(tmpA, selfn, tmpB);
      scatter_agg  <<<cdiv(N_EDGES * 16, thr), thr, 0, stream>>>(srcIdx, dstIdx, norm, tmpA, tmpB, N_EDGES);
      relu_bias_bf16<<<cdiv(N_NODES * HID, thr), thr, 0, stream>>>(tmpB, b1, h1bf);

      KSrcs g;
      g.p[0] = h1bf; g.coloff[0] = 0;
      g.p[1] = h1bf; g.coloff[1] = 32;
      for (int s = 2; s < 6; ++s) { g.p[s] = nullptr; g.coloff[s] = 0; }
      gemm_bf16_wmma<2, 0><<<dim3(gemmGrid, 1), 256, 0, stream>>>(
          g, W2p, nullptr, tmpA, nullptr, N_NODES, 0, 0);

      selfloop_init<<<cdiv(N_NODES * HID, thr), thr, 0, stream>>>(tmpA, selfn, tmpB);
      scatter_agg  <<<cdiv(N_EDGES * 16, thr), thr, 0, stream>>>(srcIdx, dstIdx, norm, tmpA, tmpB, N_EDGES);
      relu_bias_bf16<<<cdiv(N_NODES * HID, thr), thr, 0, stream>>>(
          tmpB, b2, spat + (size_t)(b * 3 + ti) * NH);
    }
  }

  // --- temporal conv1 at output times tau=10 (j=0) and tau=11 (j=1), all b at once ---
  for (int j = 0; j < 2; ++j) {
    KSrcs g;                                           // K = 3 taps * 64
    for (int s = 0; s < 6; ++s) {
      int tap = s >> 1;
      int ti = j + tap;                                // j=1, tap=2 -> SAME zero pad
      g.p[s] = (ti < 3) ? (spat + (size_t)ti * NH) : nullptr;
      g.coloff[s] = (s & 1) * 32;
    }
    gemm_bf16_wmma<6, 1><<<dim3(gemmGrid, BATCH), 256, 0, stream>>>(
        g, Wc1p, tc1b, nullptr, c1 + (size_t)j * NH, N_NODES, 3 * NH, 2 * NH);
  }

  // --- temporal conv2, only output time tau=11 (taps 0,1; tap 2 is pad), all b at once ---
  {
    KSrcs g;                                           // K = 2 taps * 64
    for (int s = 0; s < 4; ++s) {
      int tap = s >> 1;                                // 0 -> tau=10 (j0), 1 -> tau=11 (j1)
      g.p[s] = c1 + (size_t)tap * NH;
      g.coloff[s] = (s & 1) * 32;
    }
    g.p[4] = g.p[5] = nullptr; g.coloff[4] = g.coloff[5] = 0;
    gemm_bf16_wmma<4, 2><<<dim3(gemmGrid, BATCH), 256, 0, stream>>>(
        g, Wc2p, tc2b, h2f, nullptr, N_NODES, 2 * NH, NH);
  }

  // --- output projection: (B*N, 64) @ (64, 1) + bias ---
  proj_kernel<<<cdiv(BATCH * N_NODES, thr), thr, 0, stream>>>(
      h2f, outw, outb, out, BATCH * N_NODES);
}
